// TorchCausalGroupedQueryPageAttention_57870389346730
// MI455X (gfx1250) — matrix operand order repaired and checked by reading, not compiled, not run on hardware
//
#include <hip/hip_runtime.h>

typedef __bf16 bf16_t;
typedef __attribute__((ext_vector_type(16))) __bf16 v16bf;
typedef __attribute__((ext_vector_type(8)))  float  v8f;

#define NUM_SEQS        8
#define Q_LEN           4
#define N_QO_HEADS      32
#define N_KV_HEADS      8
#define GQA             4
#define HEAD_DIM        128
#define BLOCKS_PER_SEQ  256
#define KV_LEN          4096
#define NSPLIT          16
#define SPLIT_KEYS      (KV_LEN / NSPLIT)   // 256 keys per split
#define ITERS           (SPLIT_KEYS / 32)   // 8 iterations of 32 keys
#define PART_STRIDE     (16 * HEAD_DIM + 32) // floats per partial: O(16x128) + m(16) + l(16)

// ---------------------------------------------------------------------------
// Kernel 1: per-(b, kv-head, split) partial flash attention over 256 keys.
// One wave32 per split; 4 waves (4 splits) per 128-thread block.
// ---------------------------------------------------------------------------
__global__ __launch_bounds__(128) void pa_partial_kernel(
    const float* __restrict__ Q,     // [B*4, 32, 128]
    const float* __restrict__ Kc,    // [2048, 16, 8, 128]
    const float* __restrict__ Vc,    // [2048, 16, 8, 128]
    const int*   __restrict__ BT,    // [8, 256]
    float*       __restrict__ WS)
{
  const int lane = threadIdx.x & 31;
  const int wave = threadIdx.x >> 5;
  const int lo   = lane & 15;        // column / row-in-half index
  const int hi   = lane >> 4;        // which 16-lane half

  int idx = blockIdx.x;              // B * HKV * (NSPLIT/4) = 256 blocks
  const int sgrp = idx & 3;  idx >>= 2;
  const int h    = idx & 7;
  const int b    = idx >> 3;
  const int split = sgrp * 4 + wave;

  __shared__ bf16_t Psh[4][16][40];  // per-wave P staging, padded to kill bank conflicts
  bf16_t (*P)[40] = Psh[wave];

  // ---- Q tile (16 rows = q*4+g, 128 d) in A-operand layout, 4 chunks of K=32
  v16bf Qa[4];
  const float* qrow =
      Q + ((size_t)((b * Q_LEN + (lo >> 2)) * N_QO_HEADS + h * GQA + (lo & 3))) * HEAD_DIM;
  #pragma unroll
  for (int c = 0; c < 4; ++c) {
    const int d0 = c * 32 + hi * 16;
    #pragma unroll
    for (int e = 0; e < 8; ++e) {
      Qa[c][e]     = (bf16_t)qrow[d0 + e];
      Qa[c][e + 8] = (bf16_t)qrow[d0 + 8 + e];
    }
  }

  float m_i[8], l_i[8], sf[8];
  v8f O[8];
  #pragma unroll
  for (int i = 0; i < 8; ++i) { m_i[i] = -1e30f; l_i[i] = 0.0f; }
  #pragma unroll
  for (int c = 0; c < 8; ++c) {
    #pragma unroll
    for (int i = 0; i < 8; ++i) O[c][i] = 0.0f;
  }

  const float sc = 0.08838834764831845f * 1.4426950408889634f; // 1/sqrt(128) * log2(e)

  for (int it = 0; it < ITERS; ++it) {
    const int key0 = split * SPLIT_KEYS + it * 32;
    const int blk0 = BT[b * BLOCKS_PER_SEQ + (key0 >> 4)];
    const int blk1 = BT[b * BLOCKS_PER_SEQ + (key0 >> 4) + 1];

    // ---- S = Q K^T : two 16x16 score tiles via 8 bf16 WMMAs
    v8f s0 = {0.f,0.f,0.f,0.f,0.f,0.f,0.f,0.f};
    v8f s1 = {0.f,0.f,0.f,0.f,0.f,0.f,0.f,0.f};
    {
      const float* krow0 = Kc + ((size_t)((blk0 * 16 + lo) * N_KV_HEADS + h)) * HEAD_DIM;
      const float* krow1 = Kc + ((size_t)((blk1 * 16 + lo) * N_KV_HEADS + h)) * HEAD_DIM;
      #pragma unroll
      for (int c = 0; c < 4; ++c) {
        const int d0 = c * 32 + hi * 16;        // same k order as Qa: lanes 16-31 hold contraction 16..31
        v16bf Kb0, Kb1;
        #pragma unroll
        for (int e = 0; e < 16; ++e) {
          Kb0[e] = (bf16_t)krow0[d0 + e];
          Kb1[e] = (bf16_t)krow1[d0 + e];
        }
        s0 = __builtin_amdgcn_wmma_f32_16x16x32_bf16(false, Qa[c], false, Kb0, (short)0, s0, false, false);
        s1 = __builtin_amdgcn_wmma_f32_16x16x32_bf16(false, Qa[c], false, Kb1, (short)0, s1, false, false);
      }
    }

    // ---- flash softmax update (D-layout: VGPR i <-> row M=i+8*hi, lane lo <-> column)
    #pragma unroll
    for (int i = 0; i < 8; ++i) {
      const int M  = i + 8 * hi;
      const int qq = M >> 2;                    // row = q*4 + g
      float v0 = s0[i] * sc;
      float v1 = s1[i] * sc;
      v0 = (key0 + lo      > 4092 + qq) ? -1e30f : v0;   // causal mask
      v1 = (key0 + 16 + lo > 4092 + qq) ? -1e30f : v1;
      float mx = fmaxf(v0, v1);
      mx = fmaxf(mx, __shfl_xor(mx, 1, 32));
      mx = fmaxf(mx, __shfl_xor(mx, 2, 32));
      mx = fmaxf(mx, __shfl_xor(mx, 4, 32));
      mx = fmaxf(mx, __shfl_xor(mx, 8, 32));
      const float mn = fmaxf(m_i[i], mx);
      const float p0 = exp2f(v0 - mn);
      const float p1 = exp2f(v1 - mn);
      float rs = p0 + p1;
      rs += __shfl_xor(rs, 1, 32);
      rs += __shfl_xor(rs, 2, 32);
      rs += __shfl_xor(rs, 4, 32);
      rs += __shfl_xor(rs, 8, 32);
      sf[i]  = exp2f(m_i[i] - mn);
      l_i[i] = l_i[i] * sf[i] + rs;
      m_i[i] = mn;
      s0[i] = p0;
      s1[i] = p1;
    }
    #pragma unroll
    for (int c = 0; c < 8; ++c) {
      #pragma unroll
      for (int i = 0; i < 8; ++i) O[c][i] *= sf[i];
    }

    // ---- P: D-layout -> A-layout via LDS round trip
    #pragma unroll
    for (int i = 0; i < 8; ++i) {
      P[i + 8 * hi][lo]      = (bf16_t)s0[i];
      P[i + 8 * hi][16 + lo] = (bf16_t)s1[i];
    }
    asm volatile("s_wait_dscnt 0x0" ::: "memory");
    v16bf Pa;
    {
      const int k0 = hi * 16;
      #pragma unroll
      for (int e = 0; e < 8; ++e) {
        Pa[e]     = P[lo][k0 + e];
        Pa[e + 8] = P[lo][k0 + 8 + e];
      }
    }
    asm volatile("s_wait_dscnt 0x0" ::: "memory");

    // ---- O += P V : 8 bf16 WMMAs (contraction over 32 keys)
    const int blkv = hi ? blk1 : blk0;
    const float* vbase = Vc + ((size_t)(blkv * 16 * N_KV_HEADS + h)) * HEAD_DIM;
    #pragma unroll
    for (int c = 0; c < 8; ++c) {
      const float* vp = vbase + c * 16 + lo;    // column d = c*16 + lo
      v16bf Vb;
      #pragma unroll
      for (int j = 0; j < 16; ++j)              // coalesced 64B across each half-wave
        Vb[j] = (bf16_t)vp[j * (N_KV_HEADS * HEAD_DIM)];
      O[c] = __builtin_amdgcn_wmma_f32_16x16x32_bf16(false, Pa, false, Vb, (short)0, O[c], false, false);
    }
  }

  // ---- write partial (O, m, l) to workspace
  float* wp = WS + (size_t)((b * N_KV_HEADS + h) * NSPLIT + split) * PART_STRIDE;
  #pragma unroll
  for (int c = 0; c < 8; ++c) {
    #pragma unroll
    for (int i = 0; i < 8; ++i)
      wp[(i + 8 * hi) * HEAD_DIM + c * 16 + lo] = O[c][i];
  }
  if (lo == 0) {
    #pragma unroll
    for (int i = 0; i < 8; ++i) {
      wp[16 * HEAD_DIM +      i + 8 * hi] = m_i[i];
      wp[16 * HEAD_DIM + 16 + i + 8 * hi] = l_i[i];
    }
  }
}

// ---------------------------------------------------------------------------
// Kernel 2: merge NSPLIT partials per (b, kv-head) with log-sum-exp weights.
// ---------------------------------------------------------------------------
__global__ __launch_bounds__(128) void pa_merge_kernel(
    const float* __restrict__ WS, float* __restrict__ Out)
{
  const int bh = blockIdx.x;       // 64
  const int h  = bh & 7;
  const int b  = bh >> 3;
  const int d  = threadIdx.x;      // 0..127
  const float* base = WS + (size_t)bh * NSPLIT * PART_STRIDE;

  float M[16], den[16], acc[16];
  #pragma unroll
  for (int r = 0; r < 16; ++r) { M[r] = -1e30f; den[r] = 0.f; acc[r] = 0.f; }

  for (int s = 0; s < NSPLIT; ++s) {
    const float* mp = base + s * PART_STRIDE + 16 * HEAD_DIM;
    #pragma unroll
    for (int r = 0; r < 16; ++r) M[r] = fmaxf(M[r], mp[r]);
  }
  for (int s = 0; s < NSPLIT; ++s) {
    const float* pp = base + s * PART_STRIDE;
    #pragma unroll
    for (int r = 0; r < 16; ++r) {
      const float w = exp2f(pp[16 * HEAD_DIM + r] - M[r]);   // m tracked in log2 domain
      den[r] += w * pp[16 * HEAD_DIM + 16 + r];
      acc[r] += w * pp[r * HEAD_DIM + d];
    }
  }
  #pragma unroll
  for (int r = 0; r < 16; ++r) {
    const int qq = r >> 2, g = r & 3;
    Out[((size_t)(b * Q_LEN + qq) * N_QO_HEADS + h * GQA + g) * HEAD_DIM + d] = acc[r] / den[r];
  }
}

extern "C" void kernel_launch(void* const* d_in, const int* in_sizes, int n_in,
                              void* d_out, int out_size, void* d_ws, size_t ws_size,
                              hipStream_t stream) {
  const float* q  = (const float*)d_in[0];
  const float* kc = (const float*)d_in[1];
  const float* vc = (const float*)d_in[2];
  const int*   bt = (const int*)d_in[3];
  float* out = (float*)d_out;
  float* ws  = (float*)d_ws;

  // 8 seqs * 8 kv heads * 4 split-groups (4 splits/block, 1 wave32 each)
  pa_partial_kernel<<<dim3(NUM_SEQS * N_KV_HEADS * (NSPLIT / 4)), dim3(128), 0, stream>>>(
      q, kc, vc, bt, ws);
  pa_merge_kernel<<<dim3(NUM_SEQS * N_KV_HEADS), dim3(128), 0, stream>>>(ws, out);
}
